// CausalSelfAttention_42039139893449
// MI455X (gfx1250) — compile-verified, matrix operand order checked by reading, not run on hardware
//
#include <hip/hip_runtime.h>
#include <stdint.h>

// Problem constants
#define B_    4
#define S_    1024
#define DIM_  2048
#define H_    16
#define KVH_  4
#define HD_   128
#define QSZ   2048      // H*HD
#define KVSZ  512       // KVH*HD
#define NQKV  3072      // QSZ + 2*KVSZ
#define MROWS 4096      // B*S

typedef __attribute__((ext_vector_type(16))) __bf16 v16bf;
typedef __attribute__((ext_vector_type(8)))  float  v8f;

union FragA {
  v16bf v;
  uint4 q[2];
  unsigned short h[16];
};

__device__ __forceinline__ v8f v8f_zero() {
  v8f z = {0.f, 0.f, 0.f, 0.f, 0.f, 0.f, 0.f, 0.f};
  return z;
}

__device__ __forceinline__ v8f wmma_bf16(const FragA& a, const FragA& b, v8f c) {
  // D = A(16x32 bf16) * B(32x16 bf16) + C(16x16 f32)
  return __builtin_amdgcn_wmma_f32_16x16x32_bf16(
      false, a.v, false, b.v, (short)0, c, false, false);
}

// CDNA5 async DMA: global -> LDS, 16 bytes per lane, tracked by ASYNCcnt.
typedef int v4i_vs __attribute__((vector_size(16)));
__device__ __forceinline__ void async_cp16(const unsigned short* g, unsigned short* l) {
  __builtin_amdgcn_global_load_async_to_lds_b128(
      (__attribute__((address_space(1))) v4i_vs*)(g),
      (__attribute__((address_space(3))) v4i_vs*)(l), 0, 0);
}

__device__ __forceinline__ unsigned short f2bf(float f) {
  union { float f; unsigned int u; } c; c.f = f;
  unsigned int u = c.u;
  unsigned int r = u + 0x7FFFu + ((u >> 16) & 1u);  // round-to-nearest-even
  return (unsigned short)(r >> 16);
}

__device__ __forceinline__ float warp_sum(float v) {
  #pragma unroll
  for (int o = 16; o; o >>= 1) v += __shfl_xor(v, o, 32);
  return v;
}

// ---------------------------------------------------------------------------
// Ternary STE quantization (forward value), group size 64 -> bf16 weights.
// ---------------------------------------------------------------------------
__global__ __launch_bounds__(256) void quant_ternary(
    const float* __restrict__ W, unsigned short* __restrict__ Wq, int ngroups)
{
  const int g = blockIdx.x * 8 + (threadIdx.x >> 5);
  if (g >= ngroups) return;
  const int lane = threadIdx.x & 31;
  const float2 w = *(const float2*)(W + (size_t)g * 64 + lane * 2);
  const float s = warp_sum(fabsf(w.x) + fabsf(w.y));
  const float scale = fmaxf(s * (1.0f / 64.0f), 1e-8f);
  const float qa = fminf(fmaxf(rintf(w.x / scale), -1.0f), 1.0f) * scale;
  const float qb = fminf(fmaxf(rintf(w.y / scale), -1.0f), 1.0f) * scale;
  unsigned short* o = Wq + (size_t)g * 64 + lane * 2;
  o[0] = f2bf(qa);
  o[1] = f2bf(qb);
}

// ---------------------------------------------------------------------------
// f32 -> bf16 elementwise (4 elements / thread)
// ---------------------------------------------------------------------------
__global__ __launch_bounds__(256) void cvt_bf16(
    const float* __restrict__ X, unsigned short* __restrict__ Xb, int n4)
{
  const int i = blockIdx.x * 256 + threadIdx.x;
  if (i >= n4) return;
  const float4 v = *(const float4*)(X + (size_t)i * 4);
  unsigned short* o = Xb + (size_t)i * 4;
  o[0] = f2bf(v.x); o[1] = f2bf(v.y); o[2] = f2bf(v.z); o[3] = f2bf(v.w);
}

// ---------------------------------------------------------------------------
// C[M,N] = A[M,K](bf16) * B[N,K](bf16)^T, f32 accumulate via WMMA bf16.
// WG tile 128x128, 8 waves of 32x64, K-step 32.
// Double-buffered LDS, filled by GLOBAL_LOAD_ASYNC_TO_LDS_B128 (ASYNCcnt).
// ---------------------------------------------------------------------------
__global__ __launch_bounds__(256) void wmma_gemm_bf16(
    const unsigned short* __restrict__ A,
    const unsigned short* __restrict__ Bm,
    float* __restrict__ C, int M, int N, int K)
{
  __shared__ __attribute__((aligned(16))) unsigned short lA[2][128 * 32];
  __shared__ __attribute__((aligned(16))) unsigned short lB[2][128 * 32];

  const int tid  = threadIdx.x;
  const int wid  = tid >> 5;
  const int lane = tid & 31;
  const int rl   = lane & 15;
  const int hi8  = (lane >> 4) << 3;     // 0 for lanes 0-15, 8 for lanes 16-31
  const int bm   = blockIdx.x * 128;
  const int bn   = blockIdx.y * 128;
  const int mOff = (wid & 3) * 32;
  const int nOff = (wid >> 2) * 64;

  v8f acc[2][4];
  #pragma unroll
  for (int i = 0; i < 2; ++i)
    #pragma unroll
    for (int j = 0; j < 4; ++j) acc[i][j] = v8f_zero();

  const int row = tid >> 1;              // 0..127
  const int col = (tid & 1) * 16;        // 0 or 16 (bf16 elems)

  // Issue one 32-wide K-slice of A and B into LDS buffer `buf` (4 async ops).
  auto stage = [&](int kk, int buf) {
    const unsigned short* ga = A + (size_t)(bm + row) * K + kk + col;
    unsigned short* la = &lA[buf][row * 32 + col];
    async_cp16(ga, la);
    async_cp16(ga + 8, la + 8);
    const unsigned short* gb = Bm + (size_t)(bn + row) * K + kk + col;
    unsigned short* lb = &lB[buf][row * 32 + col];
    async_cp16(gb, lb);
    async_cp16(gb + 8, lb + 8);
  };

  const int niter = K >> 5;
  stage(0, 0);

  for (int i = 0; i < niter; ++i) {
    const int buf = i & 1;
    if (i + 1 < niter) {
      stage((i + 1) << 5, buf ^ 1);
      // tile i+1 (4 ops) still in flight; tile i done when ASYNCcnt <= 4
      asm volatile("s_wait_asynccnt 0x4" ::: "memory");
    } else {
      asm volatile("s_wait_asynccnt 0x0" ::: "memory");
    }
    __syncthreads();

    FragA fa[2], fb[4];
    #pragma unroll
    for (int fm = 0; fm < 2; ++fm) {
      const unsigned short* p = &lA[buf][(mOff + fm * 16 + rl) * 32 + hi8];
      fa[fm].q[0] = *(const uint4*)p;
      fa[fm].q[1] = *(const uint4*)(p + 16);
    }
    #pragma unroll
    for (int fn = 0; fn < 4; ++fn) {
      const unsigned short* p = &lB[buf][(nOff + fn * 16 + rl) * 32 + hi8];
      fb[fn].q[0] = *(const uint4*)p;
      fb[fn].q[1] = *(const uint4*)(p + 16);
    }
    #pragma unroll
    for (int fm = 0; fm < 2; ++fm)
      #pragma unroll
      for (int fn = 0; fn < 4; ++fn)
        acc[fm][fn] = wmma_bf16(fa[fm], fb[fn], acc[fm][fn]);
    __syncthreads();   // all waves done reading `buf` before it is re-staged
  }

  #pragma unroll
  for (int fm = 0; fm < 2; ++fm)
    #pragma unroll
    for (int fn = 0; fn < 4; ++fn) {
      const int n  = bn + nOff + fn * 16 + rl;
      const int mb = bm + mOff + fm * 16 + hi8;
      #pragma unroll
      for (int r = 0; r < 8; ++r)
        C[(size_t)(mb + r) * N + n] = acc[fm][fn][r];
    }
}

// ---------------------------------------------------------------------------
// Per-head RMS-norm + RoPE (+ gain) on one 128-wide head, one wave.
// ---------------------------------------------------------------------------
__device__ __forceinline__ void norm_rope_head(
    const float* __restrict__ src, int lane, int pos, float gain, float out[4])
{
  const int d0 = lane * 4;
  const float4 xv = *(const float4*)(src + d0);
  float x[4] = {xv.x, xv.y, xv.z, xv.w};
  float ss = x[0]*x[0] + x[1]*x[1] + x[2]*x[2] + x[3]*x[3];
  ss = warp_sum(ss);
  const float inv = rsqrtf(ss * (1.0f / 128.0f) + 1.1920929e-7f);
  float xn[4];
  #pragma unroll
  for (int j = 0; j < 4; ++j) xn[j] = x[j] * inv;
  float other[4];
  #pragma unroll
  for (int j = 0; j < 4; ++j) other[j] = __shfl_xor(xn[j], 16, 32);
  const float fb = (float)((lane & 15) * 4);     // rotary frequency index base
  #pragma unroll
  for (int j = 0; j < 4; ++j) {
    // inv_freq[i] = 10000^(-i/64) = exp(-i * ln(10000)/64)
    const float fr = (float)pos * __expf(-(fb + (float)j) * 0.14391157f);
    float sn, c;
    __sincosf(fr, &sn, &c);
    out[j] = ((lane < 16) ? (xn[j] * c + other[j] * sn)
                          : (xn[j] * c - other[j] * sn)) * gain;
  }
}

// qkv f32 [B*S, 3072] -> Q bf16 [B][H][S][128], K/V bf16 [B][KVH][S][128]
__global__ __launch_bounds__(256) void qkv_post(
    const float* __restrict__ qkv, const float* __restrict__ q_gain,
    unsigned short* __restrict__ Qb, unsigned short* __restrict__ Kb,
    unsigned short* __restrict__ Vb)
{
  const int idx = blockIdx.x;
  const int b = idx >> 10, s = idx & 1023;
  const float* row = qkv + (size_t)idx * NQKV;
  const int wid = threadIdx.x >> 5, lane = threadIdx.x & 31;
  const int d0 = lane * 4;
  float out[4];

  for (int hh = wid; hh < H_; hh += 8) {
    norm_rope_head(row + hh * HD_, lane, s, q_gain[hh], out);
    unsigned short* o = Qb + ((size_t)(b * H_ + hh) * S_ + s) * HD_ + d0;
    #pragma unroll
    for (int j = 0; j < 4; ++j) o[j] = f2bf(out[j]);
  }
  if (wid < 4) {
    norm_rope_head(row + QSZ + wid * HD_, lane, s, 1.0f, out);
    unsigned short* o = Kb + ((size_t)(b * KVH_ + wid) * S_ + s) * HD_ + d0;
    #pragma unroll
    for (int j = 0; j < 4; ++j) o[j] = f2bf(out[j]);
  } else {
    const int kvh = wid - 4;
    const float4 xv = *(const float4*)(row + QSZ + KVSZ + kvh * HD_ + d0);
    unsigned short* o = Vb + ((size_t)(b * KVH_ + kvh) * S_ + s) * HD_ + d0;
    o[0] = f2bf(xv.x); o[1] = f2bf(xv.y); o[2] = f2bf(xv.z); o[3] = f2bf(xv.w);
  }
}

// ---------------------------------------------------------------------------
// Differential causal flash attention. 1 WG per (b, h, 64 q-rows); 4 waves
// of 16 q-rows. K tile staged via async-to-LDS DMA; V transposed in-register.
// Y layout: [B][S][DIM] f32 with head-major dims (h*128 + d).
// ---------------------------------------------------------------------------
__global__ __launch_bounds__(128) void diff_attn(
    const unsigned short* __restrict__ Qb, const unsigned short* __restrict__ Kb,
    const unsigned short* __restrict__ Vb, const float* __restrict__ diff_lambda,
    float* __restrict__ Y)
{
  __shared__ __attribute__((aligned(16))) unsigned short lK[32 * 128];  // [key][dim]
  __shared__ __attribute__((aligned(16))) unsigned short lV[128 * 32];  // [dim][key]
  __shared__ __attribute__((aligned(16))) unsigned short lP[4][2][16 * 32];

  const int qb  = blockIdx.x & 15;
  const int h   = (blockIdx.x >> 4) & 15;
  const int b   = blockIdx.x >> 8;
  const int kvh = h >> 2;
  const int wid = threadIdx.x >> 5, lane = threadIdx.x & 31;
  const int rl = lane & 15, hi8 = (lane >> 4) << 3;
  const int q0 = qb * 64;
  const int qw = q0 + wid * 16;

  const unsigned short* Qp = Qb + (size_t)(b * H_ + h) * S_ * HD_;
  const unsigned short* Kp = Kb + (size_t)(b * KVH_ + kvh) * S_ * HD_;
  const unsigned short* Vp = Vb + (size_t)(b * KVH_ + kvh) * S_ * HD_;

  // Resident Q fragments for both halves (dims 0..63 and 64..127)
  FragA q1[2], q2[2];
  #pragma unroll
  for (int f = 0; f < 2; ++f) {
    const unsigned short* p = Qp + (size_t)(qw + rl) * HD_ + f * 32 + hi8;
    q1[f].q[0] = *(const uint4*)p;
    q1[f].q[1] = *(const uint4*)(p + 16);
    const unsigned short* p2 = p + 64;
    q2[f].q[0] = *(const uint4*)p2;
    q2[f].q[1] = *(const uint4*)(p2 + 16);
  }

  v8f y1[4], y2[4];
  #pragma unroll
  for (int i = 0; i < 4; ++i) { y1[i] = v8f_zero(); y2[i] = v8f_zero(); }
  float M1[8], L1[8], M2[8], L2[8];
  #pragma unroll
  for (int r = 0; r < 8; ++r) { M1[r] = -1e30f; M2[r] = -1e30f; L1[r] = 0.f; L2[r] = 0.f; }

  const int key = threadIdx.x >> 2;            // 0..31
  const int ch  = (threadIdx.x & 3) << 5;      // dim chunk base (32 elems)

  const int ktiles = (q0 + 64) >> 5;
  for (int kt = 0; kt < ktiles; ++kt) {
    const int k0 = kt << 5;
    {
      // K tile: async DMA straight into LDS (row-major [key][dim])
      const unsigned short* gk = Kp + (size_t)(k0 + key) * HD_ + ch;
      unsigned short* lk = lK + key * 128 + ch;
      async_cp16(gk,      lk);
      async_cp16(gk +  8, lk +  8);
      async_cp16(gk + 16, lk + 16);
      async_cp16(gk + 24, lk + 24);
      // V tile: transpose to [dim][key] through VGPRs
      union { uint4 q[4]; unsigned short u[32]; } vv;
      const uint4* sv = (const uint4*)(Vp + (size_t)(k0 + key) * HD_ + ch);
      vv.q[0] = sv[0]; vv.q[1] = sv[1]; vv.q[2] = sv[2]; vv.q[3] = sv[3];
      #pragma unroll
      for (int j = 0; j < 32; ++j) lV[(ch + j) * 32 + key] = vv.u[j];
      asm volatile("s_wait_asynccnt 0x0" ::: "memory");
    }
    __syncthreads();

    if (k0 <= qw + 15) {   // wave-uniform: this wave has unmasked keys here
      v8f cL1 = v8f_zero(), cH1 = v8f_zero(), cL2 = v8f_zero(), cH2 = v8f_zero();
      FragA bk;
      #pragma unroll
      for (int f = 0; f < 2; ++f) {
        const unsigned short* pL = lK + (size_t)rl * 128 + f * 32 + hi8;
        bk.q[0] = *(const uint4*)pL;       bk.q[1] = *(const uint4*)(pL + 16);
        cL1 = wmma_bf16(q1[f], bk, cL1);
        const unsigned short* pH = pL + 16 * 128;
        bk.q[0] = *(const uint4*)pH;       bk.q[1] = *(const uint4*)(pH + 16);
        cH1 = wmma_bf16(q1[f], bk, cH1);
        const unsigned short* pL2 = pL + 64;
        bk.q[0] = *(const uint4*)pL2;      bk.q[1] = *(const uint4*)(pL2 + 16);
        cL2 = wmma_bf16(q2[f], bk, cL2);
        const unsigned short* pH2 = pH + 64;
        bk.q[0] = *(const uint4*)pH2;      bk.q[1] = *(const uint4*)(pH2 + 16);
        cH2 = wmma_bf16(q2[f], bk, cH2);
      }

      const float scale = 0.125f;   // 1/sqrt(64)
      #pragma unroll
      for (int r = 0; r < 8; ++r) {
        const int m = qw + r + hi8;
        const bool okL = (k0 + rl) <= m;
        const bool okH = (k0 + 16 + rl) <= m;
        const float sL1 = okL ? cL1[r] * scale : -1e30f;
        const float sH1 = okH ? cH1[r] * scale : -1e30f;
        const float sL2 = okL ? cL2[r] * scale : -1e30f;
        const float sH2 = okH ? cH2[r] * scale : -1e30f;

        // half 1
        float mx = fmaxf(sL1, sH1);
        #pragma unroll
        for (int o = 8; o; o >>= 1) mx = fmaxf(mx, __shfl_xor(mx, o, 32));
        float mn = fmaxf(M1[r], mx);
        float pL = __expf(sL1 - mn), pH = __expf(sH1 - mn);
        float rs = pL + pH;
        #pragma unroll
        for (int o = 8; o; o >>= 1) rs += __shfl_xor(rs, o, 32);
        float al = __expf(M1[r] - mn);
        L1[r] = L1[r] * al + rs;  M1[r] = mn;
        #pragma unroll
        for (int nf = 0; nf < 4; ++nf) y1[nf][r] *= al;
        lP[wid][0][(r + hi8) * 32 + rl]      = f2bf(pL);
        lP[wid][0][(r + hi8) * 32 + 16 + rl] = f2bf(pH);

        // half 2
        mx = fmaxf(sL2, sH2);
        #pragma unroll
        for (int o = 8; o; o >>= 1) mx = fmaxf(mx, __shfl_xor(mx, o, 32));
        mn = fmaxf(M2[r], mx);
        pL = __expf(sL2 - mn); pH = __expf(sH2 - mn);
        rs = pL + pH;
        #pragma unroll
        for (int o = 8; o; o >>= 1) rs += __shfl_xor(rs, o, 32);
        al = __expf(M2[r] - mn);
        L2[r] = L2[r] * al + rs;  M2[r] = mn;
        #pragma unroll
        for (int nf = 0; nf < 4; ++nf) y2[nf][r] *= al;
        lP[wid][1][(r + hi8) * 32 + rl]      = f2bf(pL);
        lP[wid][1][(r + hi8) * 32 + 16 + rl] = f2bf(pH);
      }

      // Same-wave LDS RAW: make P stores visible before fragment reload.
      asm volatile("s_wait_dscnt 0" ::: "memory");

      FragA p1, p2;
      { const unsigned short* pp = &lP[wid][0][rl * 32 + hi8];
        p1.q[0] = *(const uint4*)pp;  p1.q[1] = *(const uint4*)(pp + 16); }
      { const unsigned short* pp = &lP[wid][1][rl * 32 + hi8];
        p2.q[0] = *(const uint4*)pp;  p2.q[1] = *(const uint4*)(pp + 16); }

      #pragma unroll
      for (int nf = 0; nf < 4; ++nf) {
        FragA bv;
        const unsigned short* pv = lV + (nf * 16 + rl) * 32 + hi8;
        bv.q[0] = *(const uint4*)pv;       bv.q[1] = *(const uint4*)(pv + 16);
        y1[nf] = wmma_bf16(p1, bv, y1[nf]);
        const unsigned short* pv2 = pv + 64 * 32;
        bv.q[0] = *(const uint4*)pv2;      bv.q[1] = *(const uint4*)(pv2 + 16);
        y2[nf] = wmma_bf16(p2, bv, y2[nf]);
      }
    }
    __syncthreads();
  }

  const float lam = diff_lambda[h];
  #pragma unroll
  for (int r = 0; r < 8; ++r) {
    const int srow = qw + r + hi8;
    float* o = Y + (size_t)(b * S_ + srow) * DIM_ + h * HD_;
    const float i1 = 1.0f / L1[r], i2 = 1.0f / L2[r];
    #pragma unroll
    for (int nf = 0; nf < 4; ++nf) {
      const int d = nf * 16 + rl;
      const float a  = y1[nf][r] * i1;
      const float c2 = y2[nf][r] * i2;
      o[d]      = a - lam * c2;
      o[64 + d] = a + lam * c2;
    }
  }
}

// ---------------------------------------------------------------------------
// RMS-norm over DIM=2048 per row, emit bf16.
// ---------------------------------------------------------------------------
__global__ __launch_bounds__(256) void rmsnorm_rows(
    const float* __restrict__ Yin, unsigned short* __restrict__ Yb)
{
  __shared__ float red[8];
  const int idx = blockIdx.x;
  const float* row = Yin + (size_t)idx * DIM_;
  const int t = threadIdx.x;
  const float4 a = *(const float4*)(row + t * 8);
  const float4 b = *(const float4*)(row + t * 8 + 4);
  float ss = a.x*a.x + a.y*a.y + a.z*a.z + a.w*a.w
           + b.x*b.x + b.y*b.y + b.z*b.z + b.w*b.w;
  ss = warp_sum(ss);
  if ((t & 31) == 0) red[t >> 5] = ss;
  __syncthreads();
  float tot = 0.f;
  #pragma unroll
  for (int i = 0; i < 8; ++i) tot += red[i];
  const float inv = rsqrtf(tot * (1.0f / 2048.0f) + 1.1920929e-7f);
  unsigned short* o = Yb + (size_t)idx * DIM_ + t * 8;
  o[0] = f2bf(a.x * inv); o[1] = f2bf(a.y * inv);
  o[2] = f2bf(a.z * inv); o[3] = f2bf(a.w * inv);
  o[4] = f2bf(b.x * inv); o[5] = f2bf(b.y * inv);
  o[6] = f2bf(b.z * inv); o[7] = f2bf(b.w * inv);
}

// ---------------------------------------------------------------------------
extern "C" void kernel_launch(void* const* d_in, const int* in_sizes, int n_in,
                              void* d_out, int out_size, void* d_ws, size_t ws_size,
                              hipStream_t stream)
{
  const float* x           = (const float*)d_in[0];
  const float* w_qkv       = (const float*)d_in[1];
  const float* w_proj      = (const float*)d_in[2];
  const float* q_gain      = (const float*)d_in[3];
  const float* diff_lambda = (const float*)d_in[4];
  (void)in_sizes; (void)n_in; (void)out_size; (void)ws_size;

  char* ws = (char*)d_ws;
  size_t off = 0;
  auto take = [&](size_t bytes) -> char* {
    off = (off + 255) & ~(size_t)255;
    char* p = ws + off;
    off += bytes;
    return p;
  };

  unsigned short* wqkv_b  = (unsigned short*)take((size_t)NQKV * DIM_ * 2);  // 12.6 MB
  unsigned short* wproj_b = (unsigned short*)take((size_t)DIM_ * DIM_ * 2);  //  8.4 MB
  unsigned short* xb      = (unsigned short*)take((size_t)MROWS * DIM_ * 2); // 16.8 MB (reused as yb)
  float*          qkv     = (float*)take((size_t)MROWS * NQKV * 4);          // 50.3 MB (reused as Y)
  unsigned short* Qbuf    = (unsigned short*)take((size_t)B_ * H_   * S_ * HD_ * 2);
  unsigned short* Kbuf    = (unsigned short*)take((size_t)B_ * KVH_ * S_ * HD_ * 2);
  unsigned short* Vbuf    = (unsigned short*)take((size_t)B_ * KVH_ * S_ * HD_ * 2);

  // 1) ternary-quantize weights into bf16
  quant_ternary<<<(NQKV * DIM_ / 64 + 7) / 8, 256, 0, stream>>>(w_qkv,  wqkv_b,  NQKV * DIM_ / 64);
  quant_ternary<<<(DIM_ * DIM_ / 64 + 7) / 8, 256, 0, stream>>>(w_proj, wproj_b, DIM_ * DIM_ / 64);
  // 2) x -> bf16
  cvt_bf16<<<(MROWS * DIM_ / 4 + 255) / 256, 256, 0, stream>>>(x, xb, MROWS * DIM_ / 4);
  // 3) QKV projection (WMMA, async double-buffered staging)
  wmma_gemm_bf16<<<dim3(MROWS / 128, NQKV / 128), 256, 0, stream>>>(xb, wqkv_b, qkv, MROWS, NQKV, DIM_);
  // 4) RMS-norm + RoPE + gain, head-major bf16 Q/K/V
  qkv_post<<<MROWS, 256, 0, stream>>>(qkv, q_gain, Qbuf, Kbuf, Vbuf);
  // 5) differential causal flash attention (WMMA); Y reuses qkv buffer
  float* Y = qkv;
  diff_attn<<<B_ * H_ * (S_ / 64), 128, 0, stream>>>(Qbuf, Kbuf, Vbuf, diff_lambda, Y);
  // 6) RMS-norm over DIM -> bf16 (reuses xb)
  rmsnorm_rows<<<MROWS, 256, 0, stream>>>(Y, xb);
  // 7) output projection (WMMA) -> d_out (f32)
  wmma_gemm_bf16<<<dim3(MROWS / 128, DIM_ / 128), 256, 0, stream>>>(xb, wproj_b, (float*)d_out, MROWS, DIM_, DIM_);
}
